// SwinTransformerBlock_29652454211862
// MI455X (gfx1250) — matrix operand order repaired and checked by reading, not compiled, not run on hardware
//
#include <hip/hip_runtime.h>
#include <cstddef>

typedef __attribute__((ext_vector_type(16))) _Float16 v16h;
typedef __attribute__((ext_vector_type(8)))  float    v8f;
typedef __attribute__((ext_vector_type(4)))  unsigned int v4u;
typedef __attribute__((ext_vector_type(8)))  int      v8i;
typedef __attribute__((ext_vector_type(4)))  int      v4i;

#define Bb     8
#define Hh     128
#define Wwid   128
#define Cc     256
#define HEADS  8
#define HD     32
#define WS_    8
#define SHIFT_ 4
#define NTOK   64
#define NWX    16
#define NWIN   2048

// ---------------- LDS layout (units: halves) ----------------
#define LDX       264                 // window tile row stride (64 x 256 data)
#define XW_OFF    0
#define XW_HALVES (64 * LDX)          // 16896
#define LQ        40                  // q / k row stride (64 x 32 data)
#define LV        72                  // vT row stride (32 x 64 data)
#define LP        72                  // softmax-prob row stride (64 x 64), overlays q+k
#define HEAD_HALVES (2 * 64 * LQ + 32 * LV)                // 7424
#define LDS_HALVES  (XW_HALVES + HEADS * HEAD_HALVES)      // 76288
#define LDS_BYTES   (LDS_HALVES * 2)                       // 152576 B (2 WG / WGP)

// ---------------- workspace layout (bytes) ----------------
#define SX_OFF  ((size_t)0)                                   // f16 [B,H,W,C]
#define SX_BYTES ((size_t)Bb * Hh * Wwid * Cc * 2)            // 67,108,864
#define DW_OFF  (SX_OFF + SX_BYTES)                           // f16 [B,H,W,C]
#define DW_BYTES SX_BYTES
#define WQ_OFF  (DW_OFF + DW_BYTES)                           // f16 [768,256]
#define WQ_BYTES ((size_t)3 * Cc * Cc * 2)
#define WP_OFF  (WQ_OFF + WQ_BYTES)                           // f16 [256,256]
#define WP_BYTES ((size_t)Cc * Cc * 2)

__device__ inline v8f vzero8() {
  v8f z;
#pragma unroll
  for (int i = 0; i < 8; ++i) z[i] = 0.0f;
  return z;
}

__device__ inline v8f wmma16(v16h a, v16h b, v8f c) {
  // D = A(16x32 f16) * B(32x16 f16) + C(16x16 f32)
  return __builtin_amdgcn_wmma_f32_16x16x32_f16(false, a, false, b,
                                                (short)0, c, false, false);
}

// A operand (16x32 f16), A row-major [16][lda], contiguous k (same order as load_Bt16):
// lane L: M = L%16 ; K = (L/16)*16 + 2v + h
__device__ inline v16h load_A16(const _Float16* A, int lda, int lane) {
  v16h a;
  const int m  = lane & 15;
  const int kb = (lane >> 4) << 4;
  const _Float16* row = A + m * lda;
#pragma unroll
  for (int v = 0; v < 8; ++v) {
    int kk = kb + (v << 1);
    a[2 * v]     = row[kk];
    a[2 * v + 1] = row[kk + 1];
  }
  return a;
}

// B operand (32x16 f16) supplied as Bt row-major [16][ldb] (Bt[n][k] = B[k][n]):
// lane L: N = L%16 ; K = (L/16)*16 + 2v + h
__device__ inline v16h load_Bt16(const _Float16* Bt, int ldb, int lane) {
  v16h b;
  const int n  = lane & 15;
  const int kb = (lane >> 4) << 4;
  const _Float16* row = Bt + n * ldb + kb;
#pragma unroll
  for (int v = 0; v < 8; ++v) {
    b[2 * v]     = row[2 * v];
    b[2 * v + 1] = row[2 * v + 1];
  }
  return b;
}

// ================= kernel 0: weight fp32 -> f16 =================
__global__ void cvt_w_kernel(const float* __restrict__ qkv_w,
                             const float* __restrict__ proj_w,
                             _Float16* __restrict__ wq,
                             _Float16* __restrict__ wp) {
  int i = blockIdx.x * 256 + threadIdx.x;
  if (i < 3 * Cc * Cc) wq[i] = (_Float16)qkv_w[i];
  if (i < Cc * Cc)     wp[i] = (_Float16)proj_w[i];
}

// ======== kernel 1: LayerNorm + cyclic shift (-4,-4) -> f16 sx ========
__global__ __launch_bounds__(256) void ln_shift_kernel(
    const float* __restrict__ x, const float* __restrict__ g,
    const float* __restrict__ bta, _Float16* __restrict__ sx) {
  const int tok  = blockIdx.x * 8 + (threadIdx.x >> 5);   // destination token
  const int lane = threadIdx.x & 31;
  const int b = tok >> 14, hw = tok & 16383;
  const int h = hw >> 7, w = hw & 127;
  const int shh = (h + SHIFT_) & (Hh - 1);
  const int sww = (w + SHIFT_) & (Wwid - 1);
  const float* src = x + (((size_t)b * (Hh * Wwid)) + shh * Wwid + sww) * Cc;

  float vals[8], sum = 0.f, sq = 0.f;
#pragma unroll
  for (int i = 0; i < 8; ++i) {
    float v = src[lane * 8 + i];
    vals[i] = v; sum += v; sq += v * v;
  }
#pragma unroll
  for (int d = 1; d < 32; d <<= 1) {
    sum += __shfl_xor(sum, d, 32);
    sq  += __shfl_xor(sq,  d, 32);
  }
  const float mu  = sum * (1.0f / Cc);
  const float var = sq * (1.0f / Cc) - mu * mu;
  const float inv = rsqrtf(var + 1e-5f);

  _Float16* dst = sx + (((size_t)b * (Hh * Wwid)) + h * Wwid + w) * Cc;
#pragma unroll
  for (int i = 0; i < 8; ++i) {
    int c = lane * 8 + i;
    dst[c] = (_Float16)((vals[i] - mu) * inv * g[c] + bta[c]);
  }
}

// ======== kernel 2: depthwise 7x7 conv on shifted map, LDS-tiled ========
__global__ __launch_bounds__(256) void dwconv_kernel(
    const _Float16* __restrict__ sx, const float* __restrict__ dw_w,
    const float* __restrict__ dw_b, _Float16* __restrict__ dwout) {
  __shared__ _Float16 tile[14 * 14 * 64];
  const int bid  = blockIdx.x;
  const int cblk = bid & 3;
  const int wxt  = (bid >> 2) & 15;
  const int wyt  = (bid >> 6) & 15;
  const int b    = bid >> 10;
  const int gh0 = wyt * 8, gw0 = wxt * 8, cb = cblk * 64;

  const int c    = threadIdx.x & 63;
  const int pgrp = threadIdx.x >> 6;            // 0..3

  // cooperative halo load: 14x14 positions, 4 groups x 49 each
#pragma unroll 7
  for (int i = 0; i < 49; ++i) {
    int p = pgrp * 49 + i;
    int y = p / 14, xx = p % 14;
    int gh = gh0 - 3 + y, gw = gw0 - 3 + xx;
    _Float16 v = (_Float16)0.0f;
    if (gh >= 0 && gh < Hh && gw >= 0 && gw < Wwid)
      v = sx[(((size_t)b * Hh + gh) * Wwid + gw) * Cc + cb + c];
    tile[(y * 14 + xx) * 64 + c] = v;
  }
  __syncthreads();

  float wreg[49];
#pragma unroll
  for (int k = 0; k < 49; ++k) wreg[k] = dw_w[(cb + c) * 49 + k];
  const float bias = dw_b[cb + c];

#pragma unroll
  for (int i = 0; i < 16; ++i) {                // 64 outputs / 4 groups
    int p = pgrp * 16 + i;
    int oy = p >> 3, ox = p & 7;
    float s = bias;
#pragma unroll
    for (int ky = 0; ky < 7; ++ky)
#pragma unroll
      for (int kx = 0; kx < 7; ++kx)
        s += (float)tile[((oy + ky) * 14 + (ox + kx)) * 64 + c] * wreg[ky * 7 + kx];
    dwout[(((size_t)b * Hh + gh0 + oy) * Wwid + gw0 + ox) * Cc + cb + c] = (_Float16)s;
  }
}

// ======== kernel 3: fused window attention (1 WG / window, 1 wave / head) ========
__global__ __launch_bounds__(256) void swin_attn_kernel(
    const _Float16* __restrict__ sx, const _Float16* __restrict__ dw,
    const _Float16* __restrict__ wqkv, const _Float16* __restrict__ wproj,
    const float* __restrict__ qkv_b, const float* __restrict__ proj_b,
    const float* __restrict__ rpb, const float* __restrict__ mask,
    const float* __restrict__ x_in, float* __restrict__ out) {
  extern __shared__ _Float16 lds[];
  const int tid  = threadIdx.x;
  const int lane = tid & 31;
  const int head = tid >> 5;                    // wave id == head
  const int widx = blockIdx.x;
  const int b  = widx >> 8;
  const int wy = (widx >> 4) & 15;
  const int wx = widx & 15;
  const int gh0 = wy * WS_, gw0 = wx * WS_;

  const int half = lane >> 4;                   // 0/1 : D-layout half
  const int nl   = lane & 15;

  const int Q_OFF = XW_HALVES + head * HEAD_HALVES;
  const int K_OFF = Q_OFF + 64 * LQ;
  const int V_OFF = Q_OFF + 2 * 64 * LQ;
  const int P_OFF = Q_OFF;                      // overlays q+k after softmax

  // ---- 1) stage window tile [64 x 256] f16 into LDS ----
  {
    int m = tid >> 2, cseg = (tid & 3) * 64;
    int gh = gh0 + (m >> 3), gw = gw0 + (m & 7);
    const _Float16* src = sx + (((size_t)b * Hh + gh) * Wwid + gw) * Cc + cseg;
    _Float16* dst = &lds[XW_OFF + m * LDX + cseg];
#pragma unroll
    for (int i = 0; i < 64; i += 8)
      *(uint4*)(dst + i) = *(const uint4*)(src + i);
  }
  __syncthreads();

  const float scale = 0.17677669529663687f;     // 32^-0.5

  // ---- 2) QKV GEMM for this head: [64,256] x [256,32] per q/k/v ----
#pragma unroll
  for (int wh = 0; wh < 3; ++wh) {
#pragma unroll
    for (int nt = 0; nt < 2; ++nt) {
      const int colBase = wh * Cc + head * HD + nt * 16;   // out-dim rows in wqkv
      v8f acc[4];
#pragma unroll
      for (int mt = 0; mt < 4; ++mt) acc[mt] = vzero8();
#pragma unroll
      for (int kt = 0; kt < 8; ++kt) {
        v16h bm = load_Bt16(wqkv + (size_t)colBase * Cc + kt * 32, Cc, lane);
#pragma unroll
        for (int mt = 0; mt < 4; ++mt) {
          v16h am = load_A16(&lds[XW_OFF + (mt * 16) * LDX + kt * 32], LDX, lane);
          acc[mt] = wmma16(am, bm, acc[mt]);
        }
      }
      const int nloc = nt * 16 + nl;            // 0..31 within head
      const float bias = qkv_b[wh * Cc + head * HD + nloc];
#pragma unroll
      for (int mt = 0; mt < 4; ++mt)
#pragma unroll
        for (int r = 0; r < 8; ++r) {
          int m = mt * 16 + half * 8 + r;
          float v = acc[mt][r] + bias;
          if (wh == 0)      lds[Q_OFF + m * LQ + nloc] = (_Float16)(v * scale);
          else if (wh == 1) lds[K_OFF + m * LQ + nloc] = (_Float16)v;
          else              lds[V_OFF + nloc * LV + m] = (_Float16)v;  // vT
        }
    }
  }
  __syncthreads();   // all waves done reading XW tile (reused below for dw + attn-out)

  // ---- 2b) TDM: async DMA of dw window tile into the (now free) XW region.
  // Overlaps the whole qk^T -> softmax -> p*v pipeline. 2D tensor:
  // dim0 = 8 tokens * 256 ch = 2048 contiguous halves, dim1 = 8 rows,
  // row stride W*C = 32768 halves; LDS pad 4 DWORDs every 128 DWORDs -> LDX=264.
  if (head == 0) {
    const unsigned long long gaddr = (unsigned long long)(const void*)(
        dw + (((size_t)b * Hh + gh0) * Wwid + gw0) * Cc);
    v4u g0;
    g0[0] = 1u;                                  // count=1 (valid user D#)
    g0[1] = (unsigned)(XW_OFF * 2);              // lds byte address
    g0[2] = (unsigned)(gaddr & 0xffffffffu);     // global_addr[31:0]
    g0[3] = (unsigned)((gaddr >> 32) & 0x01ffffffu) | (2u << 30);  // [56:32] | type=2
    v8i g1;
    g1[0] = (int)((1u << 16) |                   // data_size = 2 bytes
                  (1u << 20) |                   // pad_enable
                  (6u << 22) |                   // pad_interval: 128 DWORDs
                  (3u << 25));                   // pad_amount : 4 DWORDs
    g1[1] = (int)(2048u << 16);                  // tensor_dim0 = 2048
    g1[2] = (int)(8u << 16);                     // tensor_dim1 = 8
    g1[3] = (int)(2048u << 16);                  // tile_dim0 = 2048
    g1[4] = 8;                                   // tile_dim1 = 8, tile_dim2 = 0
    g1[5] = (int)(Wwid * Cc);                    // tensor_dim0_stride = 32768
    g1[6] = 0;
    g1[7] = 0;
    v4i z4;
#pragma unroll
    for (int i = 0; i < 4; ++i) z4[i] = 0;
#if defined(__clang_major__) && (__clang_major__ >= 23)
    v8i z8;
#pragma unroll
    for (int i = 0; i < 8; ++i) z8[i] = 0;
    __builtin_amdgcn_tensor_load_to_lds(g0, g1, z4, z4, z8, 0);
#else
    __builtin_amdgcn_tensor_load_to_lds(g0, g1, z4, z4, 0);
#endif
  }

  // ---- 3) S = q * k^T  (64x64, K=32) ----
  v8f S[4][4];
#pragma unroll
  for (int mt = 0; mt < 4; ++mt)
#pragma unroll
    for (int nt = 0; nt < 4; ++nt) S[mt][nt] = vzero8();
#pragma unroll
  for (int nt = 0; nt < 4; ++nt) {
    v16h bm = load_Bt16(&lds[K_OFF + (nt * 16) * LQ], LQ, lane);   // Bt = k[token][hd]
#pragma unroll
    for (int mt = 0; mt < 4; ++mt) {
      v16h am = load_A16(&lds[Q_OFF + (mt * 16) * LQ], LQ, lane);
      S[mt][nt] = wmma16(am, bm, S[mt][nt]);
    }
  }

  // ---- 4) + relative-position bias + shift mask ----
  const int wloc = wy * 16 + wx;                // window index within image
#pragma unroll
  for (int mt = 0; mt < 4; ++mt)
#pragma unroll
    for (int r = 0; r < 8; ++r) {
      const int m = mt * 16 + half * 8 + r;
      const int mi = m >> 3, mj = m & 7;
#pragma unroll
      for (int nt = 0; nt < 4; ++nt) {
        const int n = nt * 16 + nl;
        const int ni = n >> 3, nj = n & 7;
        const int rpi = (mi - ni + 7) * 15 + (mj - nj + 7);
        float add = rpb[rpi * HEADS + head] +
                    mask[((size_t)wloc * NTOK + m) * NTOK + n];
        S[mt][nt][r] = S[mt][nt][r] + add;
      }
    }

  // ---- 5) softmax over rows (row spans 16 lanes x 4 N-tiles, fixed vgpr) ----
#pragma unroll
  for (int mt = 0; mt < 4; ++mt)
#pragma unroll
    for (int r = 0; r < 8; ++r) {
      float mx = -3.4e38f;
#pragma unroll
      for (int nt = 0; nt < 4; ++nt) mx = fmaxf(mx, S[mt][nt][r]);
#pragma unroll
      for (int d = 1; d < 16; d <<= 1) mx = fmaxf(mx, __shfl_xor(mx, d, 32));
      float sum = 0.f;
#pragma unroll
      for (int nt = 0; nt < 4; ++nt) {
        float e = __expf(S[mt][nt][r] - mx);
        S[mt][nt][r] = e; sum += e;
      }
#pragma unroll
      for (int d = 1; d < 16; d <<= 1) sum += __shfl_xor(sum, d, 32);
      const float inv = 1.0f / sum;
      const int m = mt * 16 + half * 8 + r;
#pragma unroll
      for (int nt = 0; nt < 4; ++nt) {
        float p = S[mt][nt][r] * inv;
        lds[P_OFF + m * LP + nt * 16 + nl] = (_Float16)p;   // stage probs f16
      }
    }

  // ---- 6) O = P * V  (64x32, K=64) ----
  v8f O[4][2];
#pragma unroll
  for (int mt = 0; mt < 4; ++mt)
#pragma unroll
    for (int nt = 0; nt < 2; ++nt) O[mt][nt] = vzero8();
#pragma unroll
  for (int nt = 0; nt < 2; ++nt)
#pragma unroll
    for (int kt = 0; kt < 2; ++kt) {
      v16h bm = load_Bt16(&lds[V_OFF + (nt * 16) * LV + kt * 32], LV, lane);
#pragma unroll
      for (int mt = 0; mt < 4; ++mt) {
        v16h am = load_A16(&lds[P_OFF + (mt * 16) * LP + kt * 32], LP, lane);
        O[mt][nt] = wmma16(am, bm, O[mt][nt]);
      }
    }

  // ---- 7) wait TDM, then merge +dw in place in XW region (f16 [64][264]) ----
  if (head == 0) __builtin_amdgcn_s_wait_tensorcnt(0);
  __syncthreads();   // dw tile visible to all waves; everyone past p*v reads
#pragma unroll
  for (int mt = 0; mt < 4; ++mt)
#pragma unroll
    for (int nt = 0; nt < 2; ++nt) {
      const int c = head * HD + nt * 16 + nl;
#pragma unroll
      for (int r = 0; r < 8; ++r) {
        const int m = mt * 16 + half * 8 + r;
        float dv = (float)lds[XW_OFF + m * LDX + c];        // dw from TDM tile
        lds[XW_OFF + m * LDX + c] = (_Float16)(O[mt][nt][r] + dv);
      }
    }
  __syncthreads();

  // ---- 8) proj GEMM [64,256]x[256,256], this wave: cols head*32..+31 ----
#pragma unroll
  for (int nt = 0; nt < 2; ++nt) {
    const int colBase = head * HD + nt * 16;
    v8f acc[4];
#pragma unroll
    for (int mt = 0; mt < 4; ++mt) acc[mt] = vzero8();
#pragma unroll
    for (int kt = 0; kt < 8; ++kt) {
      v16h bm = load_Bt16(wproj + (size_t)colBase * Cc + kt * 32, Cc, lane);
#pragma unroll
      for (int mt = 0; mt < 4; ++mt) {
        v16h am = load_A16(&lds[XW_OFF + (mt * 16) * LDX + kt * 32], LDX, lane);
        acc[mt] = wmma16(am, bm, acc[mt]);
      }
    }
    const int col = colBase + nl;
    const float pb = proj_b[col];
#pragma unroll
    for (int mt = 0; mt < 4; ++mt)
#pragma unroll
      for (int r = 0; r < 8; ++r) {
        const int m = mt * 16 + half * 8 + r;
        const int gh = gh0 + (m >> 3), gw = gw0 + (m & 7);
        const int oh = (gh + SHIFT_) & (Hh - 1);   // reverse cyclic shift
        const int ow = (gw + SHIFT_) & (Wwid - 1);
        const size_t oidx = (((size_t)b * Hh + oh) * Wwid + ow) * Cc + col;
        out[oidx] = x_in[oidx] + acc[mt][r] + pb;  // + residual shortcut
      }
  }
}

extern "C" void kernel_launch(void* const* d_in, const int* in_sizes, int n_in,
                              void* d_out, int out_size, void* d_ws, size_t ws_size,
                              hipStream_t stream) {
  const float* x       = (const float*)d_in[0];
  const float* mask    = (const float*)d_in[1];
  // d_in[2] = x_size (static 128)
  const float* norm1_g = (const float*)d_in[3];
  const float* norm1_b = (const float*)d_in[4];
  const float* dw_w    = (const float*)d_in[5];
  const float* dw_b    = (const float*)d_in[6];
  const float* qkv_w   = (const float*)d_in[7];
  const float* qkv_b   = (const float*)d_in[8];
  const float* proj_w  = (const float*)d_in[9];
  const float* proj_b  = (const float*)d_in[10];
  const float* rpb     = (const float*)d_in[11];
  float* out = (float*)d_out;

  char* ws = (char*)d_ws;
  _Float16* sx = (_Float16*)(ws + SX_OFF);
  _Float16* dw = (_Float16*)(ws + DW_OFF);
  _Float16* wq = (_Float16*)(ws + WQ_OFF);
  _Float16* wp = (_Float16*)(ws + WP_OFF);

  cvt_w_kernel<<<(3 * Cc * Cc + 255) / 256, 256, 0, stream>>>(qkv_w, proj_w, wq, wp);
  ln_shift_kernel<<<(Bb * Hh * Wwid) / 8, 256, 0, stream>>>(x, norm1_g, norm1_b, sx);
  dwconv_kernel<<<Bb * 16 * 16 * 4, 256, 0, stream>>>(sx, dw_w, dw_b, dw);
  swin_attn_kernel<<<NWIN, 256, LDS_BYTES, stream>>>(sx, dw, wq, wp, qkv_b, proj_b,
                                                     rpb, mask, x, out);
}
